// HeadSemantic_35983236006251
// MI455X (gfx1250) — matrix operand order repaired and checked by reading, not compiled, not run on hardware
//
#include <hip/hip_runtime.h>
#include <hip/hip_bf16.h>

typedef __attribute__((ext_vector_type(16))) _Float16 v16h;
typedef __attribute__((ext_vector_type(8)))  float    v8f;
typedef __attribute__((ext_vector_type(4)))  float    f32x4;

#define DD 256
#define NSEG 4096

__device__ __forceinline__ v8f zero8() { v8f z = {}; return z; }

// ---------------------------------------------------------------------------
// Zero-fill fp32 buffer (pooled accumulator must be zeroed every launch).
// ---------------------------------------------------------------------------
__global__ void zero_kernel(float* __restrict__ p, int n) {
  int i = blockIdx.x * blockDim.x + threadIdx.x;
  if (i < n) p[i] = 0.0f;
}

// ---------------------------------------------------------------------------
// Swizzle a row-major fp32 [K][Ncols] weight matrix into f16 WMMA B-operand
// layout: for tile (kt, ct), lane l, element e:
//   dst[((kt*CT + ct)*32 + l)*16 + e] = W[32*kt + 8*(l/16) + (e%8) + 16*(e/8)][16*ct + (l%16)]
// ---------------------------------------------------------------------------
__global__ void swz_kernel(const float* __restrict__ W, _Float16* __restrict__ dst,
                           int K, int Ncols) {
  int idx = blockIdx.x * blockDim.x + threadIdx.x;
  int CT = Ncols >> 4;
  int entries = (K >> 5) * CT * 32;
  if (idx >= entries) return;
  int lane = idx & 31;
  int tile = idx >> 5;
  int ct = tile % CT;
  int kt = tile / CT;
  int kh = lane >> 4, nn = lane & 15;
  int col = ct * 16 + nn;
  v16h v;
#pragma unroll
  for (int e = 0; e < 16; ++e) {
    int k = kt * 32 + kh * 8 + (e & 7) + (e >> 3) * 16;
    v[e] = (_Float16)W[(size_t)k * Ncols + col];
  }
  *(v16h*)(dst + (size_t)idx * 16) = v;
}

// ---------------------------------------------------------------------------
// Per-segment row counts via binary search over the sorted batch array.
// ---------------------------------------------------------------------------
__global__ void counts_kernel(const int* __restrict__ batch, int n,
                              float* __restrict__ cntf) {
  int s = blockIdx.x * blockDim.x + threadIdx.x;
  if (s >= NSEG) return;
  int lo = 0, hi = n;
  while (lo < hi) { int mid = (lo + hi) >> 1; if (batch[mid] < s) lo = mid + 1; else hi = mid; }
  int lb = lo;
  lo = lb; hi = n;
  while (lo < hi) { int mid = (lo + hi) >> 1; if (batch[mid] < s + 1) lo = mid + 1; else hi = mid; }
  cntf[s] = (float)(lo - lb);
}

// ---------------------------------------------------------------------------
// Stage 1: pooled[seg][:] += x @ W_in  (bias folded in later via counts).
// Block = 8 waves; each block walks a contiguous range of 16-row tiles.
// Wave w keeps B tiles for cols [32w, 32w+32) resident in 128 VGPRs.
// A row-tile is cooperatively converted fp32->f16 into LDS in A layout.
// WMMA C accumulators carry across row-tiles while the segment is unchanged.
// ---------------------------------------------------------------------------
__global__ __launch_bounds__(256) void s1_kernel(
    const float* __restrict__ x, const int* __restrict__ batch,
    const _Float16* __restrict__ winsw, float* __restrict__ pooled,
    int nrows, int tiles_per_block, int ntiles) {
  __shared__ __align__(32) _Float16 alds[2][8 * 32 * 16];  // 2 x 8KB double buffer

  const int lane = threadIdx.x & 31;
  const int w    = threadIdx.x >> 5;
  const int kh   = lane >> 4;
  const int nn   = lane & 15;
  const int n0   = 32 * w;  // first output column owned by this wave

  // Resident B tiles: ct = 2w, 2w+1 across all 8 k-tiles (16 x v16h = 128 VGPRs)
  v16h B[8][2];
#pragma unroll
  for (int kt = 0; kt < 8; ++kt)
#pragma unroll
    for (int j = 0; j < 2; ++j)
      B[kt][j] = *(const v16h*)(winsw + (size_t)((kt * 16 + (2 * w + j)) * 32 + lane) * 16);

  v8f c0 = zero8(), c1 = zero8();
  int run_seg = -1;

  auto flushRun = [&]() {
    if (run_seg >= 0) {
      float s0 = 0.f, s1 = 0.f;
#pragma unroll
      for (int r = 0; r < 8; ++r) { s0 += c0[r]; s1 += c1[r]; }
      s0 += __shfl_xor(s0, 16, 32);
      s1 += __shfl_xor(s1, 16, 32);
      if (lane < 16) {
        atomicAdd(&pooled[run_seg * DD + n0 + lane], s0);
        atomicAdd(&pooled[run_seg * DD + n0 + 16 + lane], s1);
      }
      c0 = zero8(); c1 = zero8();
    }
  };

  int t_begin = blockIdx.x * tiles_per_block;
  int t_end   = t_begin + tiles_per_block;
  if (t_end > ntiles) t_end = ntiles;

  for (int t = t_begin; t < t_end; ++t) {
    const int m0  = t * 16;
    const int buf = t & 1;

    // Cooperative A-stage: 256 threads <-> 8 k-tiles x 32 lanes, fp32 -> f16
    {
      const int skt  = threadIdx.x >> 5;
      const int l2   = threadIdx.x & 31;
      const int arow = m0 + (l2 & 15);
      const int kb   = skt * 32 + (l2 >> 4) * 8;
      v16h a = {};
      if (arow < nrows) {
        const float* rp = x + (size_t)arow * DD;
        f32x4 f0 = *(const f32x4*)(rp + kb);
        f32x4 f1 = *(const f32x4*)(rp + kb + 4);
        f32x4 f2 = *(const f32x4*)(rp + kb + 16);
        f32x4 f3 = *(const f32x4*)(rp + kb + 20);
#pragma unroll
        for (int e = 0; e < 4; ++e) {
          a[e]      = (_Float16)f0[e];
          a[e + 4]  = (_Float16)f1[e];
          a[e + 8]  = (_Float16)f2[e];
          a[e + 12] = (_Float16)f3[e];
        }
      }
      *(v16h*)(&alds[buf][(skt * 32 + l2) * 16]) = a;
    }
    __syncthreads();

    // Segment decision: batch sorted => first==last implies uniform tile
    int mlast = m0 + 15; if (mlast > nrows - 1) mlast = nrows - 1;
    const int sf = batch[m0];
    const int sl = batch[mlast];
    const bool uniform = (sf == sl);
    if (uniform) {
      if (sf != run_seg) { flushRun(); run_seg = sf; }
    } else {
      flushRun(); run_seg = -1;
    }

    // 16 WMMAs, accumulating into (possibly carried) c0/c1
#pragma unroll
    for (int kt = 0; kt < 8; ++kt) {
      v16h a = *(const v16h*)(&alds[buf][(kt * 32 + lane) * 16]);
      c0 = __builtin_amdgcn_wmma_f32_16x16x32_f16(false, a, false, B[kt][0], (short)0, c0, false, false);
      c1 = __builtin_amdgcn_wmma_f32_16x16x32_f16(false, a, false, B[kt][1], (short)0, c1, false, false);
    }

    if (!uniform) {
      // Rare path: segment boundary inside the tile -> per-row atomics.
#pragma unroll
      for (int r = 0; r < 8; ++r) {
        const int row = m0 + r + 8 * kh;
        if (row < nrows) {
          const int s = batch[row];
          atomicAdd(&pooled[s * DD + n0 + nn], c0[r]);
          atomicAdd(&pooled[s * DD + n0 + 16 + nn], c1[r]);
        }
      }
      c0 = zero8(); c1 = zero8();
    }
  }
  flushRun();
}

// ---------------------------------------------------------------------------
// Stage 2a: tmp[4096][512] = relu((pooled + cnt*b_in) @ W1 + b1)
// 512 waves: wave -> (row-tile, column half of 256 cols). C = 16 x v8f.
// ---------------------------------------------------------------------------
__global__ __launch_bounds__(256) void s2a_kernel(
    const float* __restrict__ pooled, const float* __restrict__ cntf,
    const float* __restrict__ b_in, const _Float16* __restrict__ w1sw,
    const float* __restrict__ b1, float* __restrict__ tmp) {
  const int lane = threadIdx.x & 31, w = threadIdx.x >> 5;
  const int kh = lane >> 4, nn = lane & 15;
  const int waveid = blockIdx.x * 8 + w;  // 0..511
  const int rt = waveid >> 1;
  const int ch = waveid & 1;
  const int arow = rt * 16 + nn;
  const float* rp = pooled + (size_t)arow * DD;
  const float cr = cntf[arow];

  v8f c[16];
#pragma unroll
  for (int i = 0; i < 16; ++i) c[i] = zero8();

#pragma unroll
  for (int kt = 0; kt < 8; ++kt) {
    const int kb = kt * 32 + kh * 8;
    v16h a;
#pragma unroll
    for (int e = 0; e < 8; ++e) {
      const int k0 = kb + e;
      const int k1 = kb + 16 + e;
      a[e]     = (_Float16)(rp[k0] + cr * b_in[k0]);
      a[e + 8] = (_Float16)(rp[k1] + cr * b_in[k1]);
    }
#pragma unroll
    for (int j = 0; j < 16; ++j) {
      const int ct = ch * 16 + j;
      v16h b = *(const v16h*)(w1sw + (size_t)((kt * 32 + ct) * 32 + lane) * 16);
      c[j] = __builtin_amdgcn_wmma_f32_16x16x32_f16(false, a, false, b, (short)0, c[j], false, false);
    }
  }

#pragma unroll
  for (int j = 0; j < 16; ++j) {
    const int col = (ch * 16 + j) * 16 + nn;
    const float bb = b1[col];
#pragma unroll
    for (int r = 0; r < 8; ++r) {
      const int orow = rt * 16 + r + 8 * kh;
      float v = c[j][r] + bb;
      tmp[(size_t)orow * 512 + col] = v > 0.f ? v : 0.f;
    }
  }
}

// ---------------------------------------------------------------------------
// Stage 2b: out[4096][256] = tmp @ W2 + b2.  256 waves: wave -> row-tile.
// ---------------------------------------------------------------------------
__global__ __launch_bounds__(256) void s2b_kernel(
    const float* __restrict__ tmp, const _Float16* __restrict__ w2sw,
    const float* __restrict__ b2, float* __restrict__ out) {
  const int lane = threadIdx.x & 31, w = threadIdx.x >> 5;
  const int kh = lane >> 4, nn = lane & 15;
  const int rt = blockIdx.x * 8 + w;  // 0..255
  const float* rp = tmp + (size_t)(rt * 16 + nn) * 512;

  v8f c[16];
#pragma unroll
  for (int i = 0; i < 16; ++i) c[i] = zero8();

  for (int kt = 0; kt < 16; ++kt) {
    const int kb = kt * 32 + kh * 8;
    f32x4 f0 = *(const f32x4*)(rp + kb);
    f32x4 f1 = *(const f32x4*)(rp + kb + 4);
    f32x4 f2 = *(const f32x4*)(rp + kb + 16);
    f32x4 f3 = *(const f32x4*)(rp + kb + 20);
    v16h a;
#pragma unroll
    for (int e = 0; e < 4; ++e) {
      a[e]      = (_Float16)f0[e];
      a[e + 4]  = (_Float16)f1[e];
      a[e + 8]  = (_Float16)f2[e];
      a[e + 12] = (_Float16)f3[e];
    }
#pragma unroll
    for (int ct = 0; ct < 16; ++ct) {
      v16h b = *(const v16h*)(w2sw + (size_t)((kt * 16 + ct) * 32 + lane) * 16);
      c[ct] = __builtin_amdgcn_wmma_f32_16x16x32_f16(false, a, false, b, (short)0, c[ct], false, false);
    }
  }

#pragma unroll
  for (int ct = 0; ct < 16; ++ct) {
    const int col = ct * 16 + nn;
    const float bb = b2[col];
#pragma unroll
    for (int r = 0; r < 8; ++r)
      out[(size_t)(rt * 16 + r + 8 * kh) * DD + col] = c[ct][r] + bb;
  }
}

// ---------------------------------------------------------------------------
extern "C" void kernel_launch(void* const* d_in, const int* in_sizes, int n_in,
                              void* d_out, int out_size, void* d_ws, size_t ws_size,
                              hipStream_t stream) {
  const float* x     = (const float*)d_in[0];
  const int*   batch = (const int*)d_in[1];
  const float* W_in  = (const float*)d_in[2];
  const float* b_in  = (const float*)d_in[3];
  const float* W1    = (const float*)d_in[4];
  const float* b1    = (const float*)d_in[5];
  const float* W2    = (const float*)d_in[6];
  const float* b2    = (const float*)d_in[7];
  float* out = (float*)d_out;

  const int n = in_sizes[0] / DD;  // number of x rows (1,000,000)

  // Workspace layout
  char* ws = (char*)d_ws;
  float*     pooled = (float*)ws;                                        // 4 MB
  float*     tmp    = (float*)(ws + (size_t)NSEG * DD * 4);              // 8 MB
  _Float16*  winsw  = (_Float16*)(ws + (size_t)NSEG * DD * 4 + (size_t)NSEG * 512 * 4);
  _Float16*  w1sw   = winsw + 8 * 16 * 32 * 16;                          // 128 KB after winsw
  _Float16*  w2sw   = w1sw + 8 * 32 * 32 * 16;                           // 256 KB after w1sw
  float*     cntf   = (float*)(w2sw + 16 * 16 * 32 * 16);                // 16 KB

  // Prep: zero accumulator, swizzle weights to f16 B layout, segment counts.
  zero_kernel<<<(NSEG * DD + 255) / 256, 256, 0, stream>>>(pooled, NSEG * DD);
  swz_kernel<<<(4096 + 255) / 256, 256, 0, stream>>>(W_in, winsw, 256, 256);
  swz_kernel<<<(8192 + 255) / 256, 256, 0, stream>>>(W1, w1sw, 256, 512);
  swz_kernel<<<(8192 + 255) / 256, 256, 0, stream>>>(W2, w2sw, 512, 256);
  counts_kernel<<<(NSEG + 255) / 256, 256, 0, stream>>>(batch, n, cntf);

  // Stage 1: fused GEMM + sorted segment pooling.
  const int ntiles = (n + 15) / 16;
  const int blocks = 1024;
  const int tpb = (ntiles + blocks - 1) / blocks;
  s1_kernel<<<blocks, 256, 0, stream>>>(x, batch, winsw, pooled, n, tpb, ntiles);

  // Stage 2: pooled MLP.
  s2a_kernel<<<64, 256, 0, stream>>>(pooled, cntf, b_in, w1sw, b1, tmp);
  s2b_kernel<<<32, 256, 0, stream>>>(tmp, w2sw, b2, out);
}
